// RNAFM_SIPRED_2_38981123178940
// MI455X (gfx1250) — matrix operand order repaired and checked by reading, not compiled, not run on hardware
//
#include <hip/hip_runtime.h>
#include <hip/hip_bf16.h>

typedef __attribute__((ext_vector_type(16))) _Float16 v16h;
typedef __attribute__((ext_vector_type(8)))  _Float16 v8h;
typedef __attribute__((ext_vector_type(8)))  float    v8f;
typedef __attribute__((ext_vector_type(4)))  int      v4i;

#define GAS __attribute__((address_space(1)))
#define LAS __attribute__((address_space(3)))

#if defined(__has_builtin)
#if __has_builtin(__builtin_amdgcn_global_load_async_to_lds_b128)
#define HAVE_ASYNC_LDS 1
#endif
#endif

__device__ __forceinline__ v8f wmma_f16(v16h a, v16h b, v8f c) {
  // D = A(16x32 f16) * B(32x16 f16) + C(16x16 f32)
  return __builtin_amdgcn_wmma_f32_16x16x32_f16(false, a, false, b, (short)0, c, false, false);
}

__device__ __forceinline__ v8f zero8() {
  v8f z = {0.f,0.f,0.f,0.f,0.f,0.f,0.f,0.f};
  return z;
}

__device__ __forceinline__ void load16_f32_to_h(const float* __restrict__ p, v16h& dst) {
  const float4* q = reinterpret_cast<const float4*>(p);
#pragma unroll
  for (int j = 0; j < 4; ++j) {
    float4 t = q[j];
    dst[4*j + 0] = (_Float16)t.x;
    dst[4*j + 1] = (_Float16)t.y;
    dst[4*j + 2] = (_Float16)t.z;
    dst[4*j + 3] = (_Float16)t.w;
  }
}

// ---------------------------------------------------------------------------
// Projection: out[M, DOUT] = in[M,640] @ W[DOUT,640]^T + bias + posenc(S,DOUT)
// One 16-row tile per wave; f32->f16 in-register conversion feeding WMMA.
// ---------------------------------------------------------------------------
template<int S, int DOUT>
__global__ __launch_bounds__(256) void proj_kernel(
    const float* __restrict__ in, const float* __restrict__ W,
    const float* __restrict__ bias, float* __restrict__ out, int ntiles) {
  int wave = (blockIdx.x * blockDim.x + threadIdx.x) >> 5;
  if (wave >= ntiles) return;               // wave-uniform
  int lane = threadIdx.x & 31;
  int lo = lane & 15;
  bool hi = lane >= 16;
  int row0 = wave * 16;
  const float* arp = in + (long)(row0 + lo) * 640;
  int kb0 = hi ? 16 : 0;

  v8f acc = zero8();
#pragma unroll 2
  for (int kc = 0; kc < 640; kc += 32) {
    if (kc + 64 < 640) __builtin_prefetch(arp + kc + 64 + kb0, 0, 3);
    v16h a, b;
    // A tile: row = row0 + lo; lane half selects K sub-range kb0..kb0+15
    {
      const float4* p0 = reinterpret_cast<const float4*>(arp + kc + kb0);
      const float4* p1 = reinterpret_cast<const float4*>(arp + kc + kb0 + 8);
      float4 t0 = p0[0], t1 = p0[1], t2 = p1[0], t3 = p1[1];
      a[0]=(_Float16)t0.x; a[1]=(_Float16)t0.y; a[2]=(_Float16)t0.z; a[3]=(_Float16)t0.w;
      a[4]=(_Float16)t1.x; a[5]=(_Float16)t1.y; a[6]=(_Float16)t1.z; a[7]=(_Float16)t1.w;
      a[8]=(_Float16)t2.x; a[9]=(_Float16)t2.y; a[10]=(_Float16)t2.z; a[11]=(_Float16)t2.w;
      a[12]=(_Float16)t3.x; a[13]=(_Float16)t3.y; a[14]=(_Float16)t3.z; a[15]=(_Float16)t3.w;
    }
    // B tile: col n = lo; lo-half lanes K=kc..kc+15, hi-half K=kc+16..kc+31
    if (lo < DOUT) {
      load16_f32_to_h(W + (long)lo * 640 + kc + (hi ? 16 : 0), b);
    } else {
#pragma unroll
      for (int i = 0; i < 16; ++i) b[i] = (_Float16)0.f;
    }
    acc = wmma_f16(a, b, acc);
  }

  // D layout: lane<16 holds D[r][lo] r=0..7 ; lane>=16 holds D[8+r][lo]
  int col = lo;
  if (col < DOUT) {
    float bs = bias[col];
    float div = __expf(-(float)(col & ~1) * (9.210340371976184f / (float)DOUT));
#pragma unroll
    for (int r = 0; r < 8; ++r) {
      int row = row0 + (hi ? 8 : 0) + r;
      int pos = row % S;
      float ang = (float)pos * div;
      float pe = (col & 1) ? __cosf(ang) : __sinf(ang);
      out[(long)row * DOUT + col] = acc[r] + bs + pe;
    }
  }
}

// ---------------------------------------------------------------------------
// One transformer encoder layer (post-norm). One workgroup per sample.
// Attention/qkv/LN in f32 VALU (tiny); FFN (d->2048->d) via fused WMMA:
// GEMM1 -> bias+ReLU -> 1KB LDS transpose -> GEMM2, h never touches HBM.
// ---------------------------------------------------------------------------
template<int S, int D, int NH, int MT>
__global__ __launch_bounds__(256) void encoder_layer_kernel(
    float* __restrict__ xbuf,
    const float* __restrict__ qkv_w, const float* __restrict__ qkv_b,
    const float* __restrict__ ow,    const float* __restrict__ ob,
    const float* __restrict__ f1w,   const float* __restrict__ f1b,
    const float* __restrict__ f2w,   const float* __restrict__ f2b,
    const float* __restrict__ g1,    const float* __restrict__ b1,
    const float* __restrict__ g2,    const float* __restrict__ b2) {
  __shared__ alignas(16) float xs[MT*16][D];   // padded activations (pad rows = 0)
  __shared__ float qkvs[S][3*D];
  __shared__ float att[NH][S][S];
  __shared__ float ovs[S][D];
  __shared__ float o2[S][D];
  __shared__ float mbuf[S], vbuf[S];
  __shared__ float hacc[MT*16][16];            // FFN cross-wave f32 accumulator
  __shared__ alignas(16) _Float16 ht[8][16][32]; // per-wave h-tile transpose buffers

  const int tid = threadIdx.x;
  const int dh = D / NH;
  float* xg = xbuf + (long)blockIdx.x * S * D;

  // ---- load x (async DMA to LDS when available), zero FFN accumulator ----
#if defined(HAVE_ASYNC_LDS)
  for (int i = tid * 4; i < S * D; i += 1024) {
    __builtin_amdgcn_global_load_async_to_lds_b128(
        (GAS v4i*)(xg + i), (LAS v4i*)(&xs[0][0] + i), 0, 0);
  }
  for (int i = S * D + tid; i < MT*16*D; i += 256) (&xs[0][0])[i] = 0.f;
#if __has_builtin(__builtin_amdgcn_s_wait_asynccnt)
  __builtin_amdgcn_s_wait_asynccnt(0);
#else
  asm volatile("s_wait_asynccnt 0" ::: "memory");
#endif
#else
  for (int i = tid; i < MT*16*D; i += 256) {
    int r = i / D;
    (&xs[0][0])[i] = (r < S) ? xg[i] : 0.f;
  }
#endif
  for (int i = tid; i < MT*16*16; i += 256) (&hacc[0][0])[i] = 0.f;
  __syncthreads();

  // ---- qkv = x @ qkv_w^T + qkv_b ----
  for (int i = tid; i < S*3*D; i += 256) {
    int s = i / (3*D), e = i % (3*D);
    float acc = qkv_b[e];
    for (int k = 0; k < D; ++k) acc += xs[s][k] * qkv_w[e*D + k];
    qkvs[s][e] = acc;
  }
  __syncthreads();

  // ---- scores + softmax ----
  {
    float scale = rsqrtf((float)dh);
    for (int i = tid; i < NH*S; i += 256) {
      int h = i / S, qr = i % S;
      float mx = -1e30f;
      for (int kr = 0; kr < S; ++kr) {
        float d = 0.f;
        for (int c = 0; c < dh; ++c) d += qkvs[qr][h*dh + c] * qkvs[kr][D + h*dh + c];
        d *= scale;
        att[h][qr][kr] = d;
        mx = fmaxf(mx, d);
      }
      float sum = 0.f;
      for (int kr = 0; kr < S; ++kr) { float e = __expf(att[h][qr][kr] - mx); att[h][qr][kr] = e; sum += e; }
      float inv = 1.f / sum;
      for (int kr = 0; kr < S; ++kr) att[h][qr][kr] *= inv;
    }
  }
  __syncthreads();

  // ---- o = att @ v ----
  for (int i = tid; i < S*D; i += 256) {
    int s = i / D, c = i % D, h = c / dh;
    float acc = 0.f;
    for (int kr = 0; kr < S; ++kr) acc += att[h][s][kr] * qkvs[kr][2*D + c];
    ovs[s][c] = acc;
  }
  __syncthreads();

  // ---- out proj + residual ----
  for (int i = tid; i < S*D; i += 256) {
    int s = i / D, c = i % D;
    float acc = ob[c];
    for (int j = 0; j < D; ++j) acc += ovs[s][j] * ow[c*D + j];
    o2[s][c] = xs[s][c] + acc;
  }
  __syncthreads();

  // ---- LayerNorm 1 ----
  for (int s = tid; s < S; s += 256) {
    float m = 0.f;
    for (int c = 0; c < D; ++c) m += o2[s][c];
    m /= (float)D;
    float v = 0.f;
    for (int c = 0; c < D; ++c) { float t = o2[s][c] - m; v += t * t; }
    mbuf[s] = m; vbuf[s] = rsqrtf(v / (float)D + 1e-5f);
  }
  __syncthreads();
  for (int i = tid; i < S*D; i += 256) {
    int s = i / D, c = i % D;
    xs[s][c] = (o2[s][c] - mbuf[s]) * vbuf[s] * g1[c] + b1[c];
  }
  __syncthreads();

  // ---- fused FFN via WMMA: h = relu(x@f1w^T+b1); y += h@f2w^T ----
  {
    const int wv = tid >> 5;
    const int lane = tid & 31;
    const int lo = lane & 15;
    const bool hi = lane >= 16;
    const int kb0 = hi ? 8 : 0;

    // A1 tiles: x in A-layout, K = D padded to 32 with zeros
    v16h A1[MT];
#pragma unroll
    for (int mt = 0; mt < MT; ++mt) {
      int row = mt*16 + lo;
#pragma unroll
      for (int i = 0; i < 8; ++i) {
        int k0 = 2*kb0 + i, k1 = 2*kb0 + 8 + i;
        A1[mt][i]     = (_Float16)((k0 < D) ? xs[row][k0] : 0.f);
        A1[mt][8 + i] = (_Float16)((k1 < D) ? xs[row][k1] : 0.f);
      }
    }
    v8f C2[MT];
#pragma unroll
    for (int mt = 0; mt < MT; ++mt) C2[mt] = zero8();

#pragma unroll 2
    for (int t = 0; t < 8; ++t) {
      int nb = wv * 256 + t * 32;                  // this wave's 32-wide hidden chunk
      // prefetch next chunk's weights into L2/L0 while this chunk computes
      if (t + 1 < 8) {
        int nb2 = nb + 32;
        __builtin_prefetch(f1w + (long)(nb2 + lo) * D, 0, 3);
        __builtin_prefetch(f2w + (long)lo * 2048 + nb2, 0, 3);
        __builtin_prefetch(f1b + nb2, 0, 3);
      }
      // B1 tiles (f1w^T): col n = nb+lo (a) / nb+16+lo (b); K<=D valid, rest zero
      v16h B1a, B1b;
      if (!hi) {
        const float* w1a = f1w + (long)(nb + lo) * D;
        const float* w1b = f1w + (long)(nb + 16 + lo) * D;
#pragma unroll
        for (int i = 0; i < 16; ++i) {
          B1a[i] = (_Float16)((i < D) ? w1a[i] : 0.f);
          B1b[i] = (_Float16)((i < D) ? w1b[i] : 0.f);
        }
      } else {
#pragma unroll
        for (int i = 0; i < 16; ++i) { B1a[i] = (_Float16)0.f; B1b[i] = (_Float16)0.f; }
      }
      float bia = f1b[nb + lo];
      float bib = f1b[nb + 16 + lo];
      // B2 tile (f2w^T): col n = lo, K = nb + half-wave split
      v16h B2;
      if (lo < D) {
        load16_f32_to_h(f2w + (long)lo * 2048 + nb + (hi ? 16 : 0), B2);
      } else {
#pragma unroll
        for (int i = 0; i < 16; ++i) B2[i] = (_Float16)0.f;
      }

#pragma unroll
      for (int mt = 0; mt < MT; ++mt) {
        v8f D1a = wmma_f16(A1[mt], B1a, zero8());
        v8f D1b = wmma_f16(A1[mt], B1b, zero8());
        // bias + ReLU, transpose C-layout -> A-layout through wave-private LDS
        int rr = hi ? 8 : 0;
#pragma unroll
        for (int r = 0; r < 8; ++r) {
          ht[wv][rr + r][lo]      = (_Float16)fmaxf(D1a[r] + bia, 0.f);
          ht[wv][rr + r][16 + lo] = (_Float16)fmaxf(D1b[r] + bib, 0.f);
        }
        // vectorized A-layout read-back: two ds_load_b128 per lane
        const v8h* hp = reinterpret_cast<const v8h*>(&ht[wv][lo][0]);
        v8h p0 = hp[kb0 >> 2];
        v8h p1 = hp[(kb0 >> 2) + 1];
        v16h A2;
#pragma unroll
        for (int i = 0; i < 8; ++i) { A2[i] = p0[i]; A2[8 + i] = p1[i]; }
        C2[mt] = wmma_f16(A2, B2, C2[mt]);
      }
    }
    // cross-wave K reduction via LDS float atomics (ds_add_f32)
#pragma unroll
    for (int mt = 0; mt < MT; ++mt) {
      int rr = mt*16 + (hi ? 8 : 0);
#pragma unroll
      for (int r = 0; r < 8; ++r) atomicAdd(&hacc[rr + r][lo], C2[mt][r]);
    }
  }
  __syncthreads();

  // ---- residual + bias + LayerNorm 2, write back ----
  for (int i = tid; i < S*D; i += 256) {
    int s = i / D, c = i % D;
    o2[s][c] = xs[s][c] + hacc[s][c] + f2b[c];
  }
  __syncthreads();
  for (int s = tid; s < S; s += 256) {
    float m = 0.f;
    for (int c = 0; c < D; ++c) m += o2[s][c];
    m /= (float)D;
    float v = 0.f;
    for (int c = 0; c < D; ++c) { float t = o2[s][c] - m; v += t * t; }
    mbuf[s] = m; vbuf[s] = rsqrtf(v / (float)D + 1e-5f);
  }
  __syncthreads();
  for (int i = tid; i < S*D; i += 256) {
    int s = i / D, c = i % D;
    xg[i] = (o2[s][c] - mbuf[s]) * vbuf[s] * g2[c] + b2[c];
  }
}

// ---------------------------------------------------------------------------
// ymap projections + feature concat + eval-mode BatchNorm -> cat[B,253]
// ---------------------------------------------------------------------------
__global__ void concat_kernel(
    const float* __restrict__ x1, const float* __restrict__ x2,
    const float* __restrict__ y1w, const float* __restrict__ y1b,
    const float* __restrict__ y2w, const float* __restrict__ y2b,
    const float* __restrict__ f0, const float* __restrict__ f1,
    const float* __restrict__ f2, const float* __restrict__ f3,
    const float* __restrict__ f4, const float* __restrict__ f5,
    const float* __restrict__ f6, const float* __restrict__ f7,
    const float* __restrict__ f8,
    const float* __restrict__ bn_g, const float* __restrict__ bn_b,
    float* __restrict__ cat) {
  int idx = blockIdx.x * blockDim.x + threadIdx.x;
  if (idx >= 1024 * 253) return;
  int b = idx / 253, col = idx % 253;
  float v;
  if (col < 84) {
    int s = col >> 2, j = col & 3;
    const float* xr = x1 + (long)(b * 21 + s) * 16;
    float acc = y1b[j];
    for (int k = 0; k < 16; ++k) acc += xr[k] * y1w[j * 16 + k];
    v = acc;
  } else if (col < 143) {
    int s = col - 84;
    const float* xr = x2 + (long)(b * 59 + s) * 8;
    float acc = y2b[0];
    for (int k = 0; k < 8; ++k) acc += xr[k] * y2w[k];
    v = acc;
  } else {
    int fc = col - 143;
    if      (fc < 19)  v = f0[b * 19 + fc];
    else if (fc < 20)  v = f1[b];
    else if (fc < 21)  v = f2[b];
    else if (fc < 23)  v = f3[b * 2 + (fc - 21)];
    else if (fc < 25)  v = f4[b * 2 + (fc - 23)];
    else if (fc < 89)  v = f5[b * 64 + (fc - 25)];
    else if (fc < 105) v = f6[b * 16 + (fc - 89)];
    else if (fc < 109) v = f7[b * 4 + (fc - 105)];
    else               v = f8[b];
  }
  cat[idx] = v * rsqrtf(1.f + 1e-5f) * bn_g[col] + bn_b[col];
}

__device__ __forceinline__ float leaky(float x) { return x > 0.f ? x : 0.01f * x; }

// ---------------------------------------------------------------------------
// MLP head: 253 -> 256 -> 64 -> 16 -> 1, sigmoid. One block per sample.
// ---------------------------------------------------------------------------
__global__ __launch_bounds__(256) void mlp_kernel(
    const float* __restrict__ cat,
    const float* __restrict__ d1w, const float* __restrict__ d1b,
    const float* __restrict__ d2w, const float* __restrict__ d2b,
    const float* __restrict__ d3w, const float* __restrict__ d3b,
    const float* __restrict__ d4w, const float* __restrict__ d4b,
    float* __restrict__ out) {
  __shared__ float xin[253], h1[256], h2[64], h3[16];
  int b = blockIdx.x, tid = threadIdx.x;
  if (tid < 253) xin[tid] = cat[(long)b * 253 + tid];
  __syncthreads();
  {
    float acc = d1b[tid];
    for (int k = 0; k < 253; ++k) acc += xin[k] * d1w[tid * 253 + k];
    h1[tid] = leaky(acc);
  }
  __syncthreads();
  if (tid < 64) {
    float acc = d2b[tid];
    for (int k = 0; k < 256; ++k) acc += h1[k] * d2w[tid * 256 + k];
    h2[tid] = leaky(acc);
  }
  __syncthreads();
  if (tid < 16) {
    float acc = d3b[tid];
    for (int k = 0; k < 64; ++k) acc += h2[k] * d3w[tid * 64 + k];
    h3[tid] = leaky(acc);
  }
  __syncthreads();
  if (tid == 0) {
    float acc = d4b[0];
    for (int k = 0; k < 16; ++k) acc += h3[k] * d4w[k];
    out[b] = 1.f / (1.f + __expf(-acc));
  }
}

// ---------------------------------------------------------------------------
extern "C" void kernel_launch(void* const* d_in, const int* in_sizes, int n_in,
                              void* d_out, int out_size, void* d_ws, size_t ws_size,
                              hipStream_t stream) {
  const float* rnafm = (const float*)d_in[0];
  const float* mrna  = (const float*)d_in[1];
  const float* feats[9];
  for (int i = 0; i < 9; ++i) feats[i] = (const float*)d_in[2 + i];
  const float* xmap1_w = (const float*)d_in[11];
  const float* xmap1_b = (const float*)d_in[12];
  const float* e1_qkvw = (const float*)d_in[13];
  const float* e1_qkvb = (const float*)d_in[14];
  const float* e1_ow   = (const float*)d_in[15];
  const float* e1_ob   = (const float*)d_in[16];
  const float* e1_f1w  = (const float*)d_in[17];
  const float* e1_f1b  = (const float*)d_in[18];
  const float* e1_f2w  = (const float*)d_in[19];
  const float* e1_f2b  = (const float*)d_in[20];
  const float* e1_g1   = (const float*)d_in[21];
  const float* e1_b1   = (const float*)d_in[22];
  const float* e1_g2   = (const float*)d_in[23];
  const float* e1_b2   = (const float*)d_in[24];
  const float* ymap1_w = (const float*)d_in[25];
  const float* ymap1_b = (const float*)d_in[26];
  const float* xmap2_w = (const float*)d_in[27];
  const float* xmap2_b = (const float*)d_in[28];
  const float* e2_qkvw = (const float*)d_in[29];
  const float* e2_qkvb = (const float*)d_in[30];
  const float* e2_ow   = (const float*)d_in[31];
  const float* e2_ob   = (const float*)d_in[32];
  const float* e2_f1w  = (const float*)d_in[33];
  const float* e2_f1b  = (const float*)d_in[34];
  const float* e2_f2w  = (const float*)d_in[35];
  const float* e2_f2b  = (const float*)d_in[36];
  const float* e2_g1   = (const float*)d_in[37];
  const float* e2_b1   = (const float*)d_in[38];
  const float* e2_g2   = (const float*)d_in[39];
  const float* e2_b2   = (const float*)d_in[40];
  const float* ymap2_w = (const float*)d_in[41];
  const float* ymap2_b = (const float*)d_in[42];
  const float* bn_g    = (const float*)d_in[43];
  const float* bn_b    = (const float*)d_in[44];
  const float* d1w = (const float*)d_in[45];
  const float* d1b = (const float*)d_in[46];
  const float* d2w = (const float*)d_in[47];
  const float* d2b = (const float*)d_in[48];
  const float* d3w = (const float*)d_in[49];
  const float* d3b = (const float*)d_in[50];
  const float* d4w = (const float*)d_in[51];
  const float* d4b = (const float*)d_in[52];

  float* x1  = (float*)d_ws;            // [1024, 21, 16]
  float* x2  = x1 + 1024 * 21 * 16;     // [1024, 59, 8]
  float* cat = x2 + 1024 * 59 * 8;      // [1024, 253]

  // Projections (bandwidth-bound; 210MB in, f16 WMMA, posenc fused)
  proj_kernel<21, 16><<<168, 256, 0, stream>>>(rnafm, xmap1_w, xmap1_b, x1, 1344);
  proj_kernel<59, 8><<<472, 256, 0, stream>>>(mrna, xmap2_w, xmap2_b, x2, 3776);

  // siRNA encoder: 4 layers, d=16, 2 heads, S=21 (2 M-tiles)
  for (int l = 0; l < 4; ++l) {
    encoder_layer_kernel<21, 16, 2, 2><<<1024, 256, 0, stream>>>(
        x1,
        e1_qkvw + (long)l * 48 * 16, e1_qkvb + (long)l * 48,
        e1_ow   + (long)l * 16 * 16, e1_ob   + (long)l * 16,
        e1_f1w  + (long)l * 2048 * 16, e1_f1b + (long)l * 2048,
        e1_f2w  + (long)l * 16 * 2048, e1_f2b + (long)l * 16,
        e1_g1 + (long)l * 16, e1_b1 + (long)l * 16,
        e1_g2 + (long)l * 16, e1_b2 + (long)l * 16);
  }
  // mRNA encoder: 2 layers, d=8, 1 head, S=59 (4 M-tiles)
  for (int l = 0; l < 2; ++l) {
    encoder_layer_kernel<59, 8, 1, 4><<<1024, 256, 0, stream>>>(
        x2,
        e2_qkvw + (long)l * 24 * 8, e2_qkvb + (long)l * 24,
        e2_ow   + (long)l * 8 * 8,  e2_ob   + (long)l * 8,
        e2_f1w  + (long)l * 2048 * 8, e2_f1b + (long)l * 2048,
        e2_f2w  + (long)l * 8 * 2048, e2_f2b + (long)l * 8,
        e2_g1 + (long)l * 8, e2_b1 + (long)l * 8,
        e2_g2 + (long)l * 8, e2_b2 + (long)l * 8);
  }

  concat_kernel<<<(1024 * 253 + 255) / 256, 256, 0, stream>>>(
      x1, x2, ymap1_w, ymap1_b, ymap2_w, ymap2_b,
      feats[0], feats[1], feats[2], feats[3], feats[4],
      feats[5], feats[6], feats[7], feats[8],
      bn_g, bn_b, cat);

  mlp_kernel<<<1024, 256, 0, stream>>>(cat, d1w, d1b, d2w, d2b, d3w, d3b,
                                       d4w, d4b, (float*)d_out);
}
